// S2ConvNet_original_15960098472505
// MI455X (gfx1250) — matrix operand order repaired and checked by reading, not compiled, not run on hardware
//
#include <hip/hip_runtime.h>

#define PI_D 3.14159265358979323846
#define PI_F 3.14159265358979323846f

typedef __attribute__((ext_vector_type(16))) _Float16 v16h;
typedef __attribute__((ext_vector_type(8)))  float    v8f;

// ------------------------------------------------------------------
// Workspace layout (units: floats). Two big slabs are ping-ponged.
// ------------------------------------------------------------------
constexpr size_t N_D1F  = 15*60*29;           // 26100
constexpr size_t N_D1I  = (size_t)15*30*29*29;// 378450
constexpr size_t N_D2F  = (size_t)10*30*19*19;// 108300
constexpr size_t N_D2I  = (size_t)10*20*19*19;// 72200
constexpr size_t N_Y1   = 15*29*24;           // complex
constexpr size_t N_Y2   = (size_t)10*19*19*192; // complex
constexpr size_t N_XF1  = 32*29*60;           // complex
constexpr size_t N_X1   = 32*15*29;           // complex
constexpr size_t N_PSI1 = 15*29*20;           // complex
constexpr size_t N_PSI2 = (size_t)10*19*19*20*40; // complex

constexpr size_t OFF_D1F  = 0;
constexpr size_t OFF_D1I  = OFF_D1F + N_D1F;
constexpr size_t OFF_D2F  = OFF_D1I + N_D1I;
constexpr size_t OFF_D2I  = OFF_D2F + N_D2F;
constexpr size_t OFF_WINT = OFF_D2I + N_D2I;       // 20 (padded to 32)
constexpr size_t OFF_Y1   = OFF_WINT + 32;
constexpr size_t OFF_Y2   = OFF_Y1 + 2*N_Y1;
constexpr size_t OFF_XF1  = OFF_Y2 + 2*N_Y2;
constexpr size_t OFF_X1   = OFF_XF1 + 2*N_XF1;
constexpr size_t OFF_PSI1 = OFF_X1 + 2*N_X1;
constexpr size_t OFF_PSI2 = OFF_PSI1 + 2*N_PSI1;
constexpr size_t OFF_FEAT = OFF_PSI2 + 2*N_PSI2;   // 32*40 floats
constexpr size_t OFF_S1   = OFF_FEAT + 1280;
constexpr size_t SLAB     = 34000000;              // 136 MB slab (floats)
constexpr size_t OFF_S2   = OFF_S1 + SLAB;

// ------------------------------------------------------------------
// f64 helpers for Wigner-d / Driscoll-Healy constants
// ------------------------------------------------------------------
__device__ __forceinline__ double dfact(int n) {
  double r = 1.0;
  for (int i = 2; i <= n; ++i) r *= (double)i;
  return r;
}
__device__ __forceinline__ double ipow(double x, int e) {
  double r = 1.0;
  for (int i = 0; i < e; ++i) r *= x;
  return r;
}
__device__ double wig_d(int l, int m, int n, double beta) {
  if (m < -l || m > l || n < -l || n > l) return 0.0;
  double cb = cos(0.5*beta), sb = sin(0.5*beta);
  double pref = sqrt(dfact(l+m)*dfact(l-m)*dfact(l+n)*dfact(l-n));
  int k0 = (m-n) > 0 ? (m-n) : 0;
  int k1 = (l+m) < (l-n) ? (l+m) : (l-n);
  double s = 0.0;
  for (int k = k0; k <= k1; ++k) {
    double c = ((k & 1) ? -1.0 : 1.0) /
               (dfact(k)*dfact(l+m-k)*dfact(l-n-k)*dfact(n-m+k));
    s += c * ipow(cb, 2*l+m-n-2*k) * ipow(sb, n-m+2*k);
  }
  return pref * s;
}
__device__ double dh_w(int b, int j) {
  double s = 0.0;
  for (int k = 0; k < b; ++k)
    s += sin((2.0*j+1.0)*(2.0*k+1.0)*PI_D/(4.0*b)) / (2.0*k+1.0);
  return (2.0/b) * sin(PI_D*(2.0*j+1.0)/(4.0*b)) * s;
}
__device__ __forceinline__ double soft_beta(int b, int j) {
  return PI_D*(2.0*j+1.0)/(4.0*b);
}

// ------------------------------------------------------------------
// Constant-generation kernels
// ------------------------------------------------------------------
__global__ void k_d1f(float* D1F) {            // [15][60][29]: d^l_{m,0}(beta30)*w30
  int idx = blockIdx.x*blockDim.x + threadIdx.x;
  if (idx >= 15*60*29) return;
  int l = idx/(60*29), j = (idx/29)%60, m = idx%29;
  D1F[idx] = (float)(wig_d(l, m-14, 0, soft_beta(30, j)) * dh_w(30, j));
}
__global__ void k_d1i(float* D1I) {            // [15][30][29][29]: d^l(beta15)
  int idx = blockIdx.x*blockDim.x + threadIdx.x;
  if (idx >= 15*30*29*29) return;
  int l = idx/(30*841), j = (idx/841)%30, m = (idx/29)%29, n = idx%29;
  D1I[idx] = (float)wig_d(l, m-14, n-14, soft_beta(15, j));
}
__global__ void k_y1(float2* Y1) {             // [15][29][24] complex
  int idx = blockIdx.x*blockDim.x + threadIdx.x;
  if (idx >= 15*29*24) return;
  int l = idx/(29*24), m = (idx/24)%29, g = idx%24;
  int mc = m - 14;
  double gb = (double)(g/8 + 1) * PI_D/24.0;
  double ga = 2.0*PI_D*(double)(g%8)/8.0;
  double amp = (double)(2*l+1) * wig_d(l, mc, 0, gb);
  double ph = (double)mc * ga;                 // exp(-i*mc*ga)
  Y1[idx] = make_float2((float)(amp*cos(ph)), (float)(-amp*sin(ph)));
}
__global__ void k_d2f(float* D2F) {            // [10][30][19][19]: d^l(beta15)*w15
  int idx = blockIdx.x*blockDim.x + threadIdx.x;
  if (idx >= 10*30*19*19) return;
  int l = idx/(30*361), j = (idx/361)%30, m = (idx/19)%19, n = idx%19;
  D2F[idx] = (float)(wig_d(l, m-9, n-9, soft_beta(15, j)) * dh_w(15, j));
}
__global__ void k_d2i(float* D2I) {            // [10][20][19][19]: d^l(beta10)
  int idx = blockIdx.x*blockDim.x + threadIdx.x;
  if (idx >= 10*20*19*19) return;
  int l = idx/(20*361), j = (idx/361)%20, m = (idx/19)%19, n = idx%19;
  D2I[idx] = (float)wig_d(l, m-9, n-9, soft_beta(10, j));
}
__global__ void k_y2(float2* Y2) {             // [10][19][19][192] complex
  int idx = blockIdx.x*blockDim.x + threadIdx.x;
  if (idx >= 10*19*19*192) return;
  int l = idx/(361*192), m = (idx/(19*192))%19, n = (idx/192)%19, g = idx%192;
  int mc = m - 9, nc = n - 9;
  int ib = g/64, ia = (g/8)%8, ig = g%8;
  double gb = (double)(ib+1) * PI_D/24.0;
  double ga = 2.0*PI_D*ia/8.0, gg = 2.0*PI_D*ig/8.0;
  double amp = (double)(2*l+1) * wig_d(l, mc, nc, gb);
  double ph = (double)mc*ga + (double)nc*gg;   // exp(-i*ph)
  Y2[idx] = make_float2((float)(amp*cos(ph)), (float)(-amp*sin(ph)));
}
__global__ void k_wint(float* WINT) {          // [20]: dh_weights(10)
  int j = blockIdx.x*blockDim.x + threadIdx.x;
  if (j >= 20) return;
  WINT[j] = (float)dh_w(10, j);
}

// ------------------------------------------------------------------
// Stage A: s2_conv
// ------------------------------------------------------------------
__global__ void k_s2_xf(const float* __restrict__ x, float2* __restrict__ xf1) {
  // xf1[b][m][j] = sum_a x[b,0,j,a] * exp(-2pi i a mc/60)
  int idx = blockIdx.x*blockDim.x + threadIdx.x;
  if (idx >= 32*29*60) return;
  int b = idx/(29*60), m = (idx/60)%29, j = idx%60;
  float w0 = -2.0f*PI_F*(float)(m-14)/60.0f;
  const float* xb = x + ((size_t)b*60 + j)*60;
  float sr = 0.f, si = 0.f;
  for (int a = 0; a < 60; ++a) {
    float s, c; __sincosf(w0*(float)a, &s, &c);
    float v = xb[a]; sr += v*c; si += v*s;
  }
  xf1[idx] = make_float2(sr, si);
}
__global__ void k_s2_X(const float2* __restrict__ xf1, const float* __restrict__ D1F,
                       float2* __restrict__ X1) {
  // X1[b][l][m] = sum_j xf1[b][m][j] * D1F[l][j][m]
  int idx = blockIdx.x*blockDim.x + threadIdx.x;
  if (idx >= 32*15*29) return;
  int b = idx/(15*29), l = (idx/29)%15, m = idx%29;
  float2 acc = make_float2(0.f, 0.f);
  const float2* row = xf1 + ((size_t)b*29 + m)*60;
  for (int j = 0; j < 60; ++j) {
    float d = D1F[((size_t)l*60 + j)*29 + m];
    float2 v = row[j];
    acc.x += v.x*d; acc.y += v.y*d;
  }
  X1[idx] = acc;
}
__global__ void k_s2_psi(const float* __restrict__ w1, const float2* __restrict__ Y1,
                         float2* __restrict__ psi1) {
  // psi1[l][n][o] = sum_g w1[0][o][g] * Y1[l][n][g]
  int idx = blockIdx.x*blockDim.x + threadIdx.x;
  if (idx >= 15*29*20) return;
  int l = idx/(29*20), n = (idx/20)%29, o = idx%20;
  float2 acc = make_float2(0.f, 0.f);
  const float2* yr = Y1 + ((size_t)l*29 + n)*24;
  for (int g = 0; g < 24; ++g) {
    float w = w1[o*24 + g];
    float2 y = yr[g];
    acc.x += w*y.x; acc.y += w*y.y;
  }
  psi1[idx] = acc;
}
__global__ void k_s2_fb(const float2* __restrict__ X1, const float2* __restrict__ psi1,
                        const float* __restrict__ D1I, float2* __restrict__ Fb1) {
  // Fb1[b][o][j][m][n] = sum_l X1[b,l,m]*conj(psi1[l,n,o])*D1I[l,j,m,n]
  int idx = blockIdx.x*blockDim.x + threadIdx.x;
  if (idx >= 16147200) return;
  int n = idx%29, m = (idx/29)%29, j = (idx/841)%30, o = (idx/25230)%20, b = idx/504600;
  int mc = m-14, nc = n-14;
  int am = mc < 0 ? -mc : mc, an = nc < 0 ? -nc : nc;
  int l0 = am > an ? am : an;
  float2 acc = make_float2(0.f, 0.f);
  for (int l = l0; l < 15; ++l) {
    float2 X = X1[((size_t)b*15 + l)*29 + m];
    float2 P = psi1[((size_t)l*29 + n)*20 + o];
    float d = D1I[(((size_t)l*30 + j)*29 + m)*29 + n];
    acc.x += (X.x*P.x + X.y*P.y)*d;   // X * conj(P)
    acc.y += (X.y*P.x - X.x*P.y)*d;
  }
  Fb1[idx] = acc;
}
__global__ void k_s2_ig(const float2* __restrict__ Fb1, float2* __restrict__ T2a) {
  // T2a[b][o][j][m][g] = sum_n Fb1[...,n] * exp(+2pi i (n-14) g/30)
  int idx = blockIdx.x*blockDim.x + threadIdx.x;
  if (idx >= 16704000) return;
  int g = idx%30, m = (idx/30)%29, j = (idx/870)%30, o = (idx/26100)%20, b = idx/522000;
  const float2* row = Fb1 + ((((size_t)b*20 + o)*30 + j)*29 + m)*29;
  float w0 = 2.0f*PI_F*(float)g/30.0f;
  float2 acc = make_float2(0.f, 0.f);
  for (int n = 0; n < 29; ++n) {
    float s, c; __sincosf(w0*(float)(n-14), &s, &c);
    float2 v = row[n];
    acc.x += v.x*c - v.y*s;
    acc.y += v.x*s + v.y*c;
  }
  T2a[idx] = acc;
}
__global__ void k_s2_relu(const float2* __restrict__ T2a, const float* __restrict__ b1,
                          float* __restrict__ h1) {
  // h1[b][o][j][a][g] = relu(Re sum_m T2a[b,o,j,m,g] e^{+2pi i (m-14) a/30} + b1[o])
  int idx = blockIdx.x*blockDim.x + threadIdx.x;
  if (idx >= 17280000) return;
  int g = idx%30, a = (idx/30)%30, j = (idx/900)%30, o = (idx/27000)%20, b = idx/540000;
  float w0 = 2.0f*PI_F*(float)a/30.0f;
  const float2* base = T2a + (((size_t)b*20 + o)*30 + j)*29*30;
  float acc = 0.f;
  for (int m = 0; m < 29; ++m) {
    float s, c; __sincosf(w0*(float)(m-14), &s, &c);
    float2 v = base[(size_t)m*30 + g];
    acc += v.x*c - v.y*s;
  }
  h1[idx] = fmaxf(acc + b1[o], 0.f);
}

// ------------------------------------------------------------------
// Stage B: so3_conv
// ------------------------------------------------------------------
__global__ void k_so3_fg(const float* __restrict__ h1, float2* __restrict__ xg) {
  // xg[b][f][j][a][n] = sum_g h1[b,f,j,a,g] e^{-2pi i g (n-9)/30}
  int idx = blockIdx.x*blockDim.x + threadIdx.x;
  if (idx >= 10944000) return;
  int n = idx%19, a = (idx/19)%30, j = (idx/570)%30, f = (idx/17100)%20, b = idx/342000;
  float w0 = -2.0f*PI_F*(float)(n-9)/30.0f;
  const float* row = h1 + ((((size_t)b*20 + f)*30 + j)*30 + a)*30;
  float2 acc = make_float2(0.f, 0.f);
  for (int g = 0; g < 30; ++g) {
    float s, c; __sincosf(w0*(float)g, &s, &c);
    float v = row[g]; acc.x += v*c; acc.y += v*s;
  }
  xg[idx] = acc;
}
__global__ void k_so3_fa(const float2* __restrict__ xg, float2* __restrict__ xf2) {
  // xf2[b][f][j][m][n] = sum_a xg[b,f,j,a,n] e^{-2pi i a (m-9)/30}
  int idx = blockIdx.x*blockDim.x + threadIdx.x;
  if (idx >= 6933600) return;
  int n = idx%19, m = (idx/19)%19, j = (idx/361)%30, f = (idx/10830)%20, b = idx/216600;
  float w0 = -2.0f*PI_F*(float)(m-9)/30.0f;
  const float2* base = xg + (((size_t)b*20 + f)*30 + j)*30*19;
  float2 acc = make_float2(0.f, 0.f);
  for (int a = 0; a < 30; ++a) {
    float s, c; __sincosf(w0*(float)a, &s, &c);
    float2 v = base[(size_t)a*19 + n];
    acc.x += v.x*c - v.y*s;
    acc.y += v.x*s + v.y*c;
  }
  xf2[idx] = acc;
}
__global__ void k_so3_X(const float2* __restrict__ xf2, const float* __restrict__ D2F,
                        float2* __restrict__ X2) {
  // X2[b][f][l][m][n] = sum_j xf2[b,f,j,m,n] * D2F[l,j,m,n]
  int idx = blockIdx.x*blockDim.x + threadIdx.x;
  if (idx >= 2310400) return;
  int n = idx%19, m = (idx/19)%19, l = (idx/361)%10, f = (idx/3610)%20, b = idx/72200;
  float2 acc = make_float2(0.f, 0.f);
  for (int j = 0; j < 30; ++j) {
    float d = D2F[(((size_t)l*30 + j)*19 + m)*19 + n];
    float2 v = xf2[((((size_t)b*20 + f)*30 + j)*19 + m)*19 + n];
    acc.x += v.x*d; acc.y += v.y*d;
  }
  X2[idx] = acc;
}

// ---- WMMA psi2: per-row GEMM psi2[(l,m,n),(i,o)] = sum_g Y2[(l,m,n),g]*w2[(i,o),g]
// M = 3610 rows, N = 800 cols (exact 50 tiles), K = 192 (exact 6x32 steps).
// Complex A (Y2) x real B (w2) -> 2 accumulators, f32 accumulate.
__global__ void k_so3_psi_wmma(const float* __restrict__ w2,
                               const float2* __restrict__ Y2,
                               float2* __restrict__ psi2) {
  const int tn0 = blockIdx.x * 16;   // col tile (i,o)
  const int tm0 = blockIdx.y * 16;   // row tile (l,m,n)
  const int lane = threadIdx.x;      // 1 wave

  __shared__ _Float16 Ayr[16][40], Ayi[16][40];  // padded rows: 80B stride, 16B aligned
  __shared__ _Float16 Btw[16][40];               // transposed B: [col][K]

  v8f accR = {}, accI = {};

  for (int kk = 0; kk < 192; kk += 32) {
    for (int e = lane; e < 512; e += 32) {
      int rr = e >> 5, cc = e & 31;
      int r = tm0 + rr, g = kk + cc;
      float vr = 0.f, vi = 0.f;
      if (r < 3610) { float2 y = Y2[(size_t)r*192 + g]; vr = y.x; vi = y.y; }
      Ayr[rr][cc] = (_Float16)vr; Ayi[rr][cc] = (_Float16)vi;
    }
    for (int e = lane; e < 512; e += 32) {
      int rr = e >> 4, cc = e & 15;          // rr = K(g), cc = col
      int g = kk + rr, c = tn0 + cc;         // c < 800 always
      Btw[cc][rr] = (_Float16)w2[(size_t)c*192 + g];
    }
    __syncthreads();

    v16h a_r, a_i, b_w;
    int arow = lane & 15, abase = (lane >> 4) * 16;
    #pragma unroll
    for (int e = 0; e < 16; ++e) {
      int K = abase + e;
      a_r[e] = Ayr[arow][K]; a_i[e] = Ayi[arow][K];
    }
    int bcol = lane & 15, bbase = (lane >> 4) * 16;
    #pragma unroll
    for (int e = 0; e < 16; ++e) b_w[e] = Btw[bcol][bbase + e];

    accR = __builtin_amdgcn_wmma_f32_16x16x32_f16(false, a_r, false, b_w, (short)0, accR, false, false);
    accI = __builtin_amdgcn_wmma_f32_16x16x32_f16(false, a_i, false, b_w, (short)0, accI, false, false);
    __syncthreads();
  }

  int colg = tn0 + (lane & 15);
  int rbase = tm0 + ((lane >> 4) ? 8 : 0);
  #pragma unroll
  for (int v = 0; v < 8; ++v) {
    int row = rbase + v;
    if (row < 3610)
      psi2[(size_t)row*800 + colg] = make_float2(accR[v], accI[v]);
  }
}

// ---- WMMA Z contraction: per l, Z[(b,m),(o,k)] = sum_{(i,n)} X2 * conj(psi2)
// M = 32*19 = 608 rows, N = 40*19 = 760 cols, K = 20*19 = 380.
// f16 inputs, f32 accumulate via v_wmma_f32_16x16x32_f16 (4 products/K-step).
__global__ void k_so3_z_wmma(const float2* __restrict__ X2,
                             const float2* __restrict__ psi2,
                             float2* __restrict__ Z2) {
  const int l   = blockIdx.z;
  const int tn0 = blockIdx.x * 16;   // col tile (o,k)
  const int tm0 = blockIdx.y * 16;   // row tile (b,m)
  const int lane = threadIdx.x;      // 32 threads = 1 wave

  __shared__ _Float16 Ar[16][40],  Ai[16][40];   // A: [row][K], padded
  __shared__ _Float16 Btr[16][40], Bti[16][40];  // B transposed: [col][K], padded

  v8f acc_rr = {}, acc_ii = {}, acc_ir = {}, acc_ri = {};

  for (int kk = 0; kk < 380; kk += 32) {
    // stage A tile (rows = (b,m), K = (i,n)) as f16 into LDS
    for (int e = lane; e < 512; e += 32) {
      int rr = e >> 5, cc = e & 31;
      int r = tm0 + rr, t = kk + cc;
      float vr = 0.f, vi = 0.f;
      if (r < 608 && t < 380) {
        int b = r/19, m = r%19, i = t/19, n = t%19;
        float2 v = X2[((((size_t)b*20 + i)*10 + l)*19 + m)*19 + n];
        vr = v.x; vi = v.y;
      }
      Ar[rr][cc] = (_Float16)vr; Ai[rr][cc] = (_Float16)vi;
    }
    // stage B tile transposed: Bt[col][K]
    for (int e = lane; e < 512; e += 32) {
      int rr = e >> 4, cc = e & 15;            // rr = K row, cc = col
      int t = kk + rr, c = tn0 + cc;
      float vr = 0.f, vi = 0.f;
      if (c < 760 && t < 380) {
        int o = c/19, k = c%19, i = t/19, n = t%19;
        float2 p = psi2[((((size_t)l*19 + k)*19 + n)*20 + i)*40 + o];
        vr = p.x; vi = p.y;
      }
      Btr[cc][rr] = (_Float16)vr; Bti[cc][rr] = (_Float16)vi;
    }
    __syncthreads();

    // build fragments in contiguous K order (K = 16*(lane>>4) + e, same for A and B) (all contiguous 16B chunks now)
    v16h a_r, a_i, b_r, b_i;
    int arow = lane & 15, abase = (lane >> 4) * 16;
    #pragma unroll
    for (int e = 0; e < 16; ++e) {
      int K = abase + e;
      a_r[e] = Ar[arow][K]; a_i[e] = Ai[arow][K];
    }
    int bcol = lane & 15, bbase = (lane >> 4) * 16;
    #pragma unroll
    for (int e = 0; e < 16; ++e) {
      b_r[e] = Btr[bcol][bbase + e]; b_i[e] = Bti[bcol][bbase + e];
    }

    acc_rr = __builtin_amdgcn_wmma_f32_16x16x32_f16(false, a_r, false, b_r, (short)0, acc_rr, false, false);
    acc_ii = __builtin_amdgcn_wmma_f32_16x16x32_f16(false, a_i, false, b_i, (short)0, acc_ii, false, false);
    acc_ir = __builtin_amdgcn_wmma_f32_16x16x32_f16(false, a_i, false, b_r, (short)0, acc_ir, false, false);
    acc_ri = __builtin_amdgcn_wmma_f32_16x16x32_f16(false, a_r, false, b_i, (short)0, acc_ri, false, false);
    __syncthreads();
  }

  // epilogue: Zr = rr+ii, Zi = ir-ri (X*conj(P)); C/D layout per ISA
  int colg = tn0 + (lane & 15);
  int rbase = tm0 + ((lane >> 4) ? 8 : 0);
  #pragma unroll
  for (int v = 0; v < 8; ++v) {
    int row = rbase + v;
    if (row < 608 && colg < 760) {
      int b = row/19, m = row%19, o = colg/19, k = colg%19;
      Z2[((((size_t)b*40 + o)*10 + l)*19 + m)*19 + k] =
          make_float2(acc_rr[v] + acc_ii[v], acc_ir[v] - acc_ri[v]);
    }
  }
}

__global__ void k_so3_fb(const float2* __restrict__ Z2, const float* __restrict__ D2I,
                         float2* __restrict__ Fb2) {
  // Fb2[b][o][j][m][k] = sum_l Z2[b,o,l,m,k]*D2I[l,j,m,k]
  int idx = blockIdx.x*blockDim.x + threadIdx.x;
  if (idx >= 9241600) return;
  int k = idx%19, m = (idx/19)%19, j = (idx/361)%20, o = (idx/7220)%40, b = idx/288800;
  float2 acc = make_float2(0.f, 0.f);
  for (int l = 0; l < 10; ++l) {
    float d = D2I[(((size_t)l*20 + j)*19 + m)*19 + k];
    float2 z = Z2[((((size_t)b*40 + o)*10 + l)*19 + m)*19 + k];
    acc.x += z.x*d; acc.y += z.y*d;
  }
  Fb2[idx] = acc;
}
__global__ void k_so3_ig(const float2* __restrict__ Fb2, float2* __restrict__ Tg) {
  // Tg[b][o][j][m][g] = sum_k Fb2[...,k] e^{+2pi i (k-9) g/20}
  int idx = blockIdx.x*blockDim.x + threadIdx.x;
  if (idx >= 9728000) return;
  int g = idx%20, m = (idx/20)%19, j = (idx/380)%20, o = (idx/7600)%40, b = idx/304000;
  const float2* row = Fb2 + ((((size_t)b*40 + o)*20 + j)*19 + m)*19;
  float w0 = 2.0f*PI_F*(float)g/20.0f;
  float2 acc = make_float2(0.f, 0.f);
  for (int k = 0; k < 19; ++k) {
    float s, c; __sincosf(w0*(float)(k-9), &s, &c);
    float2 v = row[k];
    acc.x += v.x*c - v.y*s;
    acc.y += v.x*s + v.y*c;
  }
  Tg[idx] = acc;
}
__global__ void k_so3_relu(const float2* __restrict__ Tg, const float* __restrict__ b2,
                           float* __restrict__ h2) {
  // h2[b][o][j][a][g] = relu(Re sum_m Tg[b,o,j,m,g] e^{+2pi i (m-9) a/20} + b2[o])
  int idx = blockIdx.x*blockDim.x + threadIdx.x;
  if (idx >= 10240000) return;
  int g = idx%20, a = (idx/20)%20, j = (idx/400)%20, o = (idx/8000)%40, b = idx/320000;
  float w0 = 2.0f*PI_F*(float)a/20.0f;
  const float2* base = Tg + (((size_t)b*40 + o)*20 + j)*19*20;
  float acc = 0.f;
  for (int m = 0; m < 19; ++m) {
    float s, c; __sincosf(w0*(float)(m-9), &s, &c);
    float2 v = base[(size_t)m*20 + g];
    acc += v.x*c - v.y*s;
  }
  h2[idx] = fmaxf(acc + b2[o], 0.f);
}

// ------------------------------------------------------------------
// Integration + linear head
// ------------------------------------------------------------------
__global__ void k_integrate(const float* __restrict__ h2, const float* __restrict__ WINT,
                            float* __restrict__ feat) {
  // feat[b][f] = (2pi/20)^2 * sum_{j,a,g} h2[b,f,j,a,g] * WINT[j]
  int bf = blockIdx.x;                 // 32*40 blocks
  const float* base = h2 + (size_t)bf*8000;
  float s = 0.f;
  for (int t = threadIdx.x; t < 8000; t += 256) s += base[t] * WINT[t/400];
  __shared__ float red[256];
  red[threadIdx.x] = s;
  __syncthreads();
  for (int st = 128; st > 0; st >>= 1) {
    if (threadIdx.x < st) red[threadIdx.x] += red[threadIdx.x + st];
    __syncthreads();
  }
  if (threadIdx.x == 0) feat[bf] = red[0] * (float)(PI_D*PI_D/100.0);
}
__global__ void k_linear(const float* __restrict__ feat, const float* __restrict__ wlin,
                         const float* __restrict__ blin, float* __restrict__ out) {
  int idx = blockIdx.x*blockDim.x + threadIdx.x;
  if (idx >= 320) return;
  int b = idx/10, c = idx%10;
  float acc = blin[c];
  for (int f = 0; f < 40; ++f) acc += feat[b*40 + f] * wlin[c*40 + f];
  out[idx] = acc;
}

// ------------------------------------------------------------------
// Launch
// ------------------------------------------------------------------
#define GRID(n) dim3((unsigned)(((n) + 255) / 256)), dim3(256)

extern "C" void kernel_launch(void* const* d_in, const int* in_sizes, int n_in,
                              void* d_out, int out_size, void* d_ws, size_t ws_size,
                              hipStream_t stream) {
  (void)in_sizes; (void)n_in; (void)out_size; (void)ws_size;
  const float* x    = (const float*)d_in[0];
  const float* w1   = (const float*)d_in[1];
  const float* b1   = (const float*)d_in[2];
  const float* w2   = (const float*)d_in[3];
  const float* b2   = (const float*)d_in[4];
  const float* wlin = (const float*)d_in[5];
  const float* blin = (const float*)d_in[6];
  float* out = (float*)d_out;

  float*  ws   = (float*)d_ws;
  float*  D1F  = ws + OFF_D1F;
  float*  D1I  = ws + OFF_D1I;
  float*  D2F  = ws + OFF_D2F;
  float*  D2I  = ws + OFF_D2I;
  float*  WINT = ws + OFF_WINT;
  float2* Y1   = (float2*)(ws + OFF_Y1);
  float2* Y2   = (float2*)(ws + OFF_Y2);
  float2* xf1  = (float2*)(ws + OFF_XF1);
  float2* X1   = (float2*)(ws + OFF_X1);
  float2* psi1 = (float2*)(ws + OFF_PSI1);
  float2* psi2 = (float2*)(ws + OFF_PSI2);
  float*  feat = ws + OFF_FEAT;
  float*  S1f  = ws + OFF_S1;  float2* S1c = (float2*)S1f;
  float*  S2f  = ws + OFF_S2;  float2* S2c = (float2*)S2f;

  // slab aliases (lifetimes ping-pong S1/S2)
  float2* Fb1 = S1c;   // 16.15M c
  float2* T2a = S2c;   // 16.70M c
  float*  h1  = S1f;   // 17.28M f
  float2* xg  = S2c;   // 10.94M c
  float2* xf2 = S1c;   //  6.93M c
  float2* X2  = S2c;   //  2.31M c
  float2* Z2  = S1c;   //  4.62M c
  float2* Fb2 = S2c;   //  9.24M c
  float2* Tg  = S1c;   //  9.73M c
  float*  h2  = S2f;   // 10.24M f

  // constants (regenerated every call; deterministic)
  k_d1f <<<GRID(15*60*29),    0, stream>>>(D1F);
  k_d1i <<<GRID(15*30*29*29), 0, stream>>>(D1I);
  k_y1  <<<GRID(15*29*24),    0, stream>>>(Y1);
  k_d2f <<<GRID(10*30*19*19), 0, stream>>>(D2F);
  k_d2i <<<GRID(10*20*19*19), 0, stream>>>(D2I);
  k_y2  <<<GRID(10*19*19*192),0, stream>>>(Y2);
  k_wint<<<GRID(20),          0, stream>>>(WINT);

  // stage A: s2_conv + relu
  k_s2_xf  <<<GRID(32*29*60), 0, stream>>>(x, xf1);
  k_s2_X   <<<GRID(32*15*29), 0, stream>>>(xf1, D1F, X1);
  k_s2_psi <<<GRID(15*29*20), 0, stream>>>(w1, Y1, psi1);
  k_s2_fb  <<<GRID(16147200), 0, stream>>>(X1, psi1, D1I, Fb1);
  k_s2_ig  <<<GRID(16704000), 0, stream>>>(Fb1, T2a);
  k_s2_relu<<<GRID(17280000), 0, stream>>>(T2a, b1, h1);

  // stage B: so3_conv + relu
  k_so3_fg  <<<GRID(10944000), 0, stream>>>(h1, xg);
  k_so3_fa  <<<GRID(6933600),  0, stream>>>(xg, xf2);
  k_so3_X   <<<GRID(2310400),  0, stream>>>(xf2, D2F, X2);
  k_so3_psi_wmma<<<dim3(50, 226), dim3(32), 0, stream>>>(w2, Y2, psi2);
  k_so3_z_wmma  <<<dim3(48, 38, 10), dim3(32), 0, stream>>>(X2, psi2, Z2);
  k_so3_fb  <<<GRID(9241600),  0, stream>>>(Z2, D2I, Fb2);
  k_so3_ig  <<<GRID(9728000),  0, stream>>>(Fb2, Tg);
  k_so3_relu<<<GRID(10240000), 0, stream>>>(Tg, b2, h2);

  // integrate + linear
  k_integrate<<<dim3(32*40), dim3(256), 0, stream>>>(h2, WINT, feat);
  k_linear   <<<GRID(320), 0, stream>>>(feat, wlin, blin, out);
}
